// CustomGraphSAGE_72232759984603
// MI455X (gfx1250) — matrix operand order repaired and checked by reading, not compiled, not run on hardware
//
#include <hip/hip_runtime.h>
#include <hip/hip_bf16.h>

#define N_NODES 100000
#define N_EDGES 1250000
#define D_FEAT  64
#define D_IN2   128   // 2*D_FEAT
#define D_OUT   64

typedef __attribute__((ext_vector_type(16))) __bf16    v16bf;
typedef __attribute__((ext_vector_type(8)))  float     v8f;
typedef __attribute__((ext_vector_type(4)))  unsigned  u32x4;

__device__ __forceinline__ unsigned pack2bf(float a, float b) {
    unsigned ua = __builtin_bit_cast(unsigned, a);
    ua += 0x7FFFu + ((ua >> 16) & 1u);                 // RNE
    unsigned ub = __builtin_bit_cast(unsigned, b);
    ub += 0x7FFFu + ((ub >> 16) & 1u);
    return (ua >> 16) | (ub & 0xFFFF0000u);
}

__device__ __forceinline__ v16bf make_frag(u32x4 lo, u32x4 hi) {
    union { u32x4 u[2]; v16bf v; } x;
    x.u[0] = lo; x.u[1] = hi;
    return x.v;
}

__device__ __forceinline__ u32x4 ld16(const __bf16* p) {
    return *(const u32x4*)p;                            // 16B-aligned global_load_b128
}

// ---------------- kernel 0: zero sums + counts workspace ----------------
__global__ void sage_zero_ws(float* __restrict__ p, int n) {
    int i = blockIdx.x * blockDim.x + threadIdx.x;
    if (i < n) p[i] = 0.0f;
}

// ---------------- kernel 1: edge scatter (copy_u + sum, degree) ---------
__global__ void sage_scatter(const float* __restrict__ h,
                             const int*   __restrict__ src,
                             const int*   __restrict__ dst,
                             float* __restrict__ sums,
                             float* __restrict__ cnts) {
    long gid = (long)blockIdx.x * blockDim.x + threadIdx.x;
    long e = gid >> 6;
    int  f = (int)(gid & 63);
    if (e >= N_EDGES) return;
    int s = src[e];
    int d = dst[e];
    float v = h[(long)s * D_FEAT + f];
    atomicAdd(&sums[(long)d * D_FEAT + f], v);
    if (f == 0) atomicAdd(&cnts[d], 1.0f);
}

// ---------------- kernel 2a: build Abf = [bf16(h) | bf16(mean)] ---------
// one thread per (node, feature-pair); packed bf16x2 stores
__global__ void sage_convert_a(const float* __restrict__ h,
                               const float* __restrict__ sums,
                               const float* __restrict__ cnts,
                               unsigned* __restrict__ Abf_u32) {  // [N_NODES][64] u32
    int gid  = blockIdx.x * blockDim.x + threadIdx.x;   // N_NODES*32 threads
    int node = gid >> 5;
    int j    = gid & 31;                                 // feature pair 0..31
    if (node >= N_NODES) return;
    int f = j * 2;
    const float* hr = h    + (long)node * D_FEAT;
    const float* sr = sums + (long)node * D_FEAT;
    float inv = 1.0f / fmaxf(cnts[node], 1.0f);          // DGL mean: 0 for isolated
    unsigned* row = Abf_u32 + (long)node * (D_IN2 / 2);
    row[j]      = pack2bf(hr[f],       hr[f + 1]);
    row[32 + j] = pack2bf(sr[f] * inv, sr[f + 1] * inv);
}

// ---------------- kernel 2b: W -> bf16 ----------------------------------
__global__ void sage_convert_w(const float* __restrict__ W,
                               unsigned* __restrict__ Wbf_u32) {  // 4096 u32
    int i = blockIdx.x * blockDim.x + threadIdx.x;
    if (i < (D_OUT * D_IN2) / 2)
        Wbf_u32[i] = pack2bf(W[2 * i], W[2 * i + 1]);
}

// ---------------- kernel 3: WMMA GEMM -----------------------------------
// block = 128 threads = 4 waves; block covers 32 nodes. Wave w computes two
// 16x16 output tiles (rows node0..+15 and node0+16..+31) for columns
// [16w, 16w+16), sharing one B fragment per k-step. 32 WMMA / block.
__launch_bounds__(128)
__global__ void sage_gemm(const __bf16* __restrict__ Abf,   // [N_NODES][128]
                          const __bf16* __restrict__ Wbf,   // [64][128] (= W^T cols)
                          const float*  __restrict__ bias,
                          float* __restrict__ out) {
    const int tid   = threadIdx.x;
    const int lane  = tid & 31;
    const int wave  = tid >> 5;
    const int node0 = blockIdx.x * 32;
    const int cbase = wave * 16;
    const int col   = lane & 15;
    const int khalf = lane >> 4;                // 0 or 1
    const int koff  = khalf * 8;

    // A fragment rows (16-bit A layout): lane holds row lane&15;
    // lanes 0-15: K {kb..kb+7, kb+16..kb+23}; lanes 16-31: +8
    const __bf16* arow0 = Abf + (long)(node0 + (lane & 15)) * D_IN2;
    const __bf16* arow1 = arow0 + 16L * D_IN2;
    // B fragment: column cbase+col = row of Wbf; lanes 0-15: K {kb..kb+7, kb+16..kb+23},
    // lanes 16-31: +8 (same K order as A)
    const __bf16* wrow  = Wbf + (long)(cbase + col) * D_IN2 + khalf * 8;

    v8f acc0 = {};
    v8f acc1 = {};
    #pragma unroll
    for (int kk = 0; kk < 4; ++kk) {
        const int kb = kk * 32;
        v16bf bfrag = make_frag(ld16(wrow + kb),             ld16(wrow + kb + 16));
        v16bf a0    = make_frag(ld16(arow0 + kb + koff),     ld16(arow0 + kb + 16 + koff));
        v16bf a1    = make_frag(ld16(arow1 + kb + koff),     ld16(arow1 + kb + 16 + koff));
        acc0 = __builtin_amdgcn_wmma_f32_16x16x32_bf16(
            false, a0, false, bfrag, (short)0, acc0, false, false);
        acc1 = __builtin_amdgcn_wmma_f32_16x16x32_bf16(
            false, a1, false, bfrag, (short)0, acc1, false, false);
    }

    const float bv = bias[cbase + col];
    #pragma unroll
    for (int v = 0; v < 8; ++v) {
        int row = v + khalf * 8;                // C/D layout: VGPR v -> M=v (+8 hi lanes)
        out[(long)(node0 + row)      * D_OUT + cbase + col] = acc0[v] + bv;
        out[(long)(node0 + 16 + row) * D_OUT + cbase + col] = acc1[v] + bv;
    }
}

extern "C" void kernel_launch(void* const* d_in, const int* in_sizes, int n_in,
                              void* d_out, int out_size, void* d_ws, size_t ws_size,
                              hipStream_t stream) {
    const float* h    = (const float*)d_in[0];
    const float* W    = (const float*)d_in[1];
    const float* b    = (const float*)d_in[2];
    const int*   src  = (const int*)d_in[3];
    const int*   dst  = (const int*)d_in[4];
    float*       out  = (float*)d_out;

    // workspace layout (all 16B-aligned):
    //   sums : N*64 f32  (25.6 MB)
    //   cnts : N    f32  ( 0.4 MB)
    //   Abf  : N*128 bf16 (25.6 MB)
    //   Wbf  : 64*128 bf16 (16 KB)
    float*   sums = (float*)d_ws;
    float*   cnts = sums + (size_t)N_NODES * D_FEAT;
    __bf16*  Abf  = (__bf16*)(cnts + N_NODES);
    __bf16*  Wbf  = Abf + (size_t)N_NODES * D_IN2;

    int nz = N_NODES * (D_FEAT + 1);
    sage_zero_ws<<<(nz + 255) / 256, 256, 0, stream>>>(sums, nz);

    long nthreads = (long)N_EDGES * 64;
    sage_scatter<<<(unsigned)((nthreads + 255) / 256), 256, 0, stream>>>(
        h, src, dst, sums, cnts);

    sage_convert_w<<<16, 256, 0, stream>>>(W, (unsigned*)Wbf);

    int na = N_NODES * 32;
    sage_convert_a<<<(na + 255) / 256, 256, 0, stream>>>(
        h, sums, cnts, (unsigned*)Abf);

    sage_gemm<<<N_NODES / 32, 128, 0, stream>>>(Abf, Wbf, b, out);

    (void)in_sizes; (void)n_in; (void)out_size; (void)ws_size;
}
